// DCTModel_87436944212528
// MI455X (gfx1250) — matrix operand order repaired and checked by reading, not compiled, not run on hardware
//
#include <hip/hip_runtime.h>
#include <cstddef>

typedef __attribute__((ext_vector_type(16))) _Float16 v16h;
typedef __attribute__((ext_vector_type(8)))  float    v8f;

#define HB 112            // blocks per side (output spatial)
#define HI 112            // input spatial
#define HO 896            // upsampled side = coeff-row count
#define KPAD 128          // K=112 padded to 128 for 4x WMMA K=32 steps

// ---------------------------------------------------------------------------
// kernel 1: fused (x*0.5)*255 scale + RGB->YCbCr mix (constants drop out:
// they only feed the DC coefficient, which is excluded from MAIN_CHANNELS)
// ---------------------------------------------------------------------------
__global__ __launch_bounds__(256) void ycbcr_kernel(const float* __restrict__ x,
                                                    float* __restrict__ z) {
    int idx = blockIdx.x * 256 + threadIdx.x;          // over 16*112*112
    if (idx >= 16 * HI * HI) return;
    int b  = idx / (HI * HI);
    int hw = idx - b * (HI * HI);
    const float* xb = x + (size_t)b * 3 * HI * HI + hw;
    float r = xb[0], g = xb[HI * HI], bl = xb[2 * HI * HI];
    float* zb = z + (size_t)b * 3 * HI * HI + hw;
    zb[0]         = 127.5f * ( 0.299f    * r + 0.587f    * g + 0.114f    * bl);
    zb[HI * HI]   = 127.5f * (-0.168736f * r - 0.331264f * g + 0.5f      * bl);
    zb[2 * HI*HI] = 127.5f * ( 0.5f      * r - 0.418688f * g - 0.081312f * bl);
}

// ---------------------------------------------------------------------------
// kernel 2: dense combined operator G[896][112]:
//   G[8i+u][p] = (alpha_u/2) * sum_x cos((x+.5)u*pi/8) * Ah[8i+x][p]
// (nscale is separable: alpha_u/2 folded here, alpha_v/2 folded in G^T)
// ---------------------------------------------------------------------------
__global__ __launch_bounds__(256) void buildG_kernel(float* __restrict__ G) {
    int idx = blockIdx.x * 256 + threadIdx.x;          // over 896*112
    if (idx >= HO * HI) return;
    int o = idx / HI, p = idx - o * HI;
    int u = o & 7;
    float acc = 0.f;
    #pragma unroll
    for (int xi = 0; xi < 8; ++xi) {
        int row = (o & ~7) + xi;                       // 8i + x
        float src = (float)row * (111.0f / 895.0f);
        int i0 = (int)src; if (i0 > 110) i0 = 110;
        float w = src - (float)i0;
        float a = (p == i0) ? (1.0f - w) : ((p == i0 + 1) ? w : 0.0f);
        float hxu = __builtin_cosf((2.f * xi + 1.f) * (float)u * (3.14159265358979f / 16.0f));
        acc += hxu * a;
    }
    G[idx] = acc * ((u == 0) ? 0.35355339059f : 0.5f); // alpha_u / 2
}

// ---------------------------------------------------------------------------
// kernel 3: G^T pre-swizzled into v_wmma_f32_16x16x32_f16 B-fragment layout.
// Fragment f = nt*4 + kt; lane L slot s -> K = s + (L>=16 ? 16 : 0),
// N = nt*16 + (L&15).  Each lane's 16 halves are contiguous (one 32B load).
// K padded 112 -> 128 with zeros.
// ---------------------------------------------------------------------------
__global__ __launch_bounds__(256) void buildGt_kernel(const float* __restrict__ G,
                                                      _Float16* __restrict__ GtSwz) {
    int idx = blockIdx.x * 256 + threadIdx.x;          // over 56*4*32*16
    if (idx >= 56 * 4 * 32 * 16) return;
    int s = idx & 15;
    int L = (idx >> 4) & 31;
    int f = idx >> 9;
    int kt = f & 3, nt = f >> 2;
    int q = kt * 32 + s + ((L & 16) ? 16 : 0);
    int n = (nt << 4) + (L & 15);
    float val = (q < HI) ? G[n * HI + q] : 0.0f;       // Gt[q][n] = G[n][q]
    GtSwz[idx] = (_Float16)val;
}

__device__ __forceinline__ int chan_of(int idx) {      // idx = u*8+v
    switch (idx) { case 0: case 1: case 2: case 3: case 4:
                   case 5: case 8: case 9: case 16: case 24: return -1; }
    int drop = 6;
    if (idx > 8)  ++drop;
    if (idx > 9)  ++drop;
    if (idx > 16) ++drop;
    if (idx > 24) ++drop;
    return idx - drop;
}

// ---------------------------------------------------------------------------
// main kernel: per WG = (b, r, 16-coeff-row stripe).
//   T = G_stripe(16x112) * z(112x112)      (VALU, K=112)
//   OUT_stripe = T(16x112,f16) * Gt(112x896,f16)  via 28 WMMA per wave
//   re-coalesced channel-scatter store via LDS
// ---------------------------------------------------------------------------
__global__ __launch_bounds__(256) void dct_main_kernel(const float* __restrict__ zAll,
                                                       const float* __restrict__ G,
                                                       const _Float16* __restrict__ GtSwz,
                                                       float* __restrict__ out) {
    __shared__ float    out_lds[16 * 8 * 113];         // [m][v][j], 57856 B (aliased as G stripe)
    __shared__ _Float16 temp_lds[16 * KPAD];           // T in f16, 4096 B

    const int t     = threadIdx.x;
    const int ipair = blockIdx.x;                      // 0..55  -> rows o0..o0+15
    const int r     = blockIdx.y;                      // YCbCr channel
    const int b     = blockIdx.z;
    const int o0    = ipair << 4;

    // ---- phase 1: stage G stripe (16x112 f32) into LDS (aliases out_lds) ----
    float* G_lds = out_lds;
    for (int e = t; e < 16 * HI; e += 256)
        G_lds[e] = G[(o0 + e / HI) * HI + (e % HI)];
    __syncthreads();

    // ---- phase 2: T[m][q] = sum_p G[o0+m][p] * z[p][q], store f16 -----------
    if (t < 224) {
        int q  = t % HI;
        int mg = t / HI;                               // 0 or 1 -> rows mg*8..mg*8+7
        float acc[8] = {0.f, 0.f, 0.f, 0.f, 0.f, 0.f, 0.f, 0.f};
        const float* zcol = zAll + (size_t)(b * 3 + r) * HI * HI + q;
        for (int p = 0; p < HI; ++p) {
            float zv = zcol[p * HI];                   // coalesced across q
            #pragma unroll
            for (int mm = 0; mm < 8; ++mm)
                acc[mm] += G_lds[(mg * 8 + mm) * HI + p] * zv;
        }
        #pragma unroll
        for (int mm = 0; mm < 8; ++mm)
            temp_lds[(mg * 8 + mm) * KPAD + q] = (_Float16)acc[mm];
    } else {
        int tt = t - 224;                              // zero K-padding 112..127
        for (int e = tt * 8; e < tt * 8 + 8; ++e)
            temp_lds[(e >> 4) * KPAD + HI + (e & 15)] = (_Float16)0.0f;
    }
    __syncthreads();

    // ---- phase 3: WMMA  OUT = T * Gt  (M=16, N=896 in 56 tiles, K=128) ------
    const int wave = t >> 5, lane = t & 31;
    const int m_a  = lane & 15;
    const int hi8  = (lane & 16) >> 1;                 // 0 / 8

    v16h afrag[4];
    #pragma unroll
    for (int kt = 0; kt < 4; ++kt) {
        #pragma unroll
        for (int s = 0; s < 16; ++s) {
            int kk = s + (hi8 << 1);                   // contiguous K = s + 16*(lane>=16), matches B
            afrag[kt][s] = temp_lds[m_a * KPAD + kt * 32 + kk];
        }
    }

    const v16h* bfrags = (const v16h*)GtSwz;
    for (int q7 = 0; q7 < 7; ++q7) {
        int nt = wave + (q7 << 3);                     // 0..55, uniform per wave
        v8f acc = {};
        #pragma unroll
        for (int kt = 0; kt < 4; ++kt) {
            v16h bf = bfrags[(nt * 4 + kt) * 32 + lane];
            acc = __builtin_amdgcn_wmma_f32_16x16x32_f16(
                      false, afrag[kt], false, bf, (short)0, acc, false, false);
        }
        int nlane = (nt << 4) + (lane & 15);
        int j = nlane >> 3, v = nlane & 7;
        #pragma unroll
        for (int g = 0; g < 8; ++g)
            out_lds[((g + hi8) * 8 + v) * 113 + j] = acc[g];
    }
    __syncthreads();

    // ---- phase 4: coalesced stores of the 54 kept (u,v) channels ------------
    for (int e = t; e < 16 * 8 * HB; e += 256) {
        int run = e / HB;                              // (m,v)
        int j   = e - run * HB;
        int m = run >> 3, v = run & 7;
        int o = o0 + m;
        int i = o >> 3, u = o & 7;
        int k = chan_of((u << 3) | v);
        if (k >= 0)
            out[(((size_t)(b * 162 + r * 54 + k)) * HB + i) * HB + j] =
                out_lds[(m * 8 + v) * 113 + j];
    }
}

// ---------------------------------------------------------------------------
extern "C" void kernel_launch(void* const* d_in, const int* in_sizes, int n_in,
                              void* d_out, int out_size, void* d_ws, size_t ws_size,
                              hipStream_t stream) {
    (void)in_sizes; (void)n_in; (void)out_size; (void)ws_size;
    const float* x   = (const float*)d_in[0];
    float*       out = (float*)d_out;

    // workspace layout (all regions fully rewritten every launch)
    char* ws = (char*)d_ws;
    float*    zAll  = (float*)ws;                                  //  2,408,448 B
    float*    G     = (float*)(ws + (size_t)16 * 3 * HI * HI * 4); //    401,408 B
    _Float16* GtSwz = (_Float16*)(ws + (size_t)16 * 3 * HI * HI * 4
                                     + (size_t)HO * HI * 4);       //    229,376 B

    ycbcr_kernel <<<(16 * HI * HI + 255) / 256, 256, 0, stream>>>(x, zAll);
    buildG_kernel<<<(HO * HI + 255) / 256,      256, 0, stream>>>(G);
    buildGt_kernel<<<(56 * 4 * 32 * 16 + 255) / 256, 256, 0, stream>>>(G, GtSwz);
    dct_main_kernel<<<dim3(56, 3, 16), 256, 0, stream>>>(zAll, G, GtSwz, out);
}
